// BaselineGNN_45586782880378
// MI455X (gfx1250) — matrix operand order repaired and checked by reading, not compiled, not run on hardware
//
#include <hip/hip_runtime.h>
#include <hip/hip_bf16.h>

typedef __attribute__((ext_vector_type(16))) __bf16 v16bf;
typedef __attribute__((ext_vector_type(8)))  float  v8f;

#define HF 128           // feature width (F == H == 128)
#define EPSBN 1e-5f

// ---------------------------------------------------------------------------
// Degree / dinv
// ---------------------------------------------------------------------------
__global__ void k_deg_init(float* __restrict__ dinv, int n) {
    int i = blockIdx.x * blockDim.x + threadIdx.x;
    if (i < n) dinv[i] = 1.0f;                    // self loop contributes 1
}
__global__ void k_deg_edges(const int* __restrict__ dst, float* __restrict__ dinv, int e) {
    int i = blockIdx.x * blockDim.x + threadIdx.x;
    if (i < e) atomicAdd(&dinv[dst[i]], 1.0f);
}
__global__ void k_deg_fin(float* __restrict__ dinv, int n) {
    int i = blockIdx.x * blockDim.x + threadIdx.x;
    if (i < n) {
        float d = dinv[i];
        dinv[i] = (d > 0.0f) ? rsqrtf(d) : 0.0f;
    }
}

// ---------------------------------------------------------------------------
// Zero fill (float4 granularity; n4 = count of float4)
// ---------------------------------------------------------------------------
__global__ void k_zero4(float4* __restrict__ p, long n4) {
    long i = (long)blockIdx.x * blockDim.x + threadIdx.x;
    if (i < n4) p[i] = make_float4(0.f, 0.f, 0.f, 0.f);
}
__global__ void k_zero(float* __restrict__ p, long n) {
    long i = (long)blockIdx.x * blockDim.x + threadIdx.x;
    if (i < n) p[i] = 0.0f;
}

// ---------------------------------------------------------------------------
// Swizzle W (128x128 row-major fp32, W[k][n] at k*128+n) into bf16 B-fragment
// order: frag index t = (ct*4+ks)*32 + lane, 16 halves each.
// Contiguous k order used here for A and B (K=32): lanes 0-15 hold K=0..15 (VGPR j -> K=2j,2j+1),
// lanes 16-31 hold K=16..31. Column N = ct*16 + (lane&15).
// ---------------------------------------------------------------------------
__global__ void k_wswz(const float* __restrict__ W, __bf16* __restrict__ Bswz) {
    int t = blockIdx.x * blockDim.x + threadIdx.x;     // 0..1023
    if (t >= 32 * 32) return;
    int lane = t & 31, tile = t >> 5;                  // tile = ct*4 + ks
    int ct = tile >> 2, ks = tile & 3;
    int half = lane >> 4, col = ct * 16 + (lane & 15);
    __bf16* d = Bswz + (long)t * 16;
    #pragma unroll
    for (int j = 0; j < 8; ++j) {
        int k = ks * 32 + half * 16 + 2 * j;
        d[2 * j]     = (__bf16)W[k * HF + col];
        d[2 * j + 1] = (__bf16)W[(k + 1) * HF + col];
    }
}

// ---------------------------------------------------------------------------
// GEMM: C[Npad x 128] = A[n x 128] @ W, via v_wmma_f32_16x16x32_bf16.
// Block = 256 threads = 8 waves; wave w covers rows [blk*128+w*16, +16) x 128.
// C is padded to a multiple of 128 rows -> stores are UNGUARDED (no branches).
// A loads clamp the row to n-1 so we never read past user input.
// ---------------------------------------------------------------------------
__global__ void k_gemm(const float* __restrict__ A, const __bf16* __restrict__ Bswz,
                       float* __restrict__ C, int nrows_data) {
    const int lane = threadIdx.x & 31;
    const int wave = threadIdx.x >> 5;
    const int r0   = blockIdx.x * 128 + wave * 16;
    const int half = lane >> 4;
    const int mrow = lane & 15;

    int arow = r0 + mrow;
    if (arow >= nrows_data) arow = nrows_data - 1;     // clamp reads only
    const float* ap = A + (long)arow * HF;

    v16bf afrag[4];
    #pragma unroll
    for (int ks = 0; ks < 4; ++ks) {
        #pragma unroll
        for (int j = 0; j < 8; ++j) {
            int k = ks * 32 + half * 16 + 2 * j;
            float2 t = *(const float2*)(ap + k);
            afrag[ks][2 * j]     = (__bf16)t.x;
            afrag[ks][2 * j + 1] = (__bf16)t.y;
        }
    }

    #pragma unroll
    for (int ct = 0; ct < 8; ++ct) {
        v8f acc = {};
        #pragma unroll
        for (int ks = 0; ks < 4; ++ks) {
            v16bf bfrag = *(const v16bf*)(Bswz + ((long)(ct * 4 + ks) * 32 + lane) * 16);
            acc = __builtin_amdgcn_wmma_f32_16x16x32_bf16(
                      false, afrag[ks], false, bfrag, (short)0, acc, false, false);
        }
        #pragma unroll
        for (int i = 0; i < 8; ++i) {
            int row = r0 + i + half * 8;               // C/D layout: M=i (+8 hi half)
            C[(long)row * HF + ct * 16 + mrow] = acc[i];   // unguarded (padded buffer)
        }
    }
}

// ---------------------------------------------------------------------------
// Edge scatter: one wave per edge; lane handles 4 consecutive features.
// Edge index forced wave-uniform so src/dst/dinv loads scalarize.
// agg[dst] += xw[src] * dinv[src]*dinv[dst]   (atomic fp32, L2-resident)
// ---------------------------------------------------------------------------
__global__ void k_scatter(const float* __restrict__ xw, const int* __restrict__ src,
                          const int* __restrict__ dst, const float* __restrict__ dinv,
                          float* __restrict__ agg, int nedges) {
    int e = __builtin_amdgcn_readfirstlane(blockIdx.x * 8 + (threadIdx.x >> 5));
    if (e >= nedges) return;
    int lane = threadIdx.x & 31;
    int s = src[e], d = dst[e];
    float norm = dinv[s] * dinv[d];
    const float4 v = *(const float4*)(xw + (long)s * HF + lane * 4);
    float* o = agg + (long)d * HF + lane * 4;
    atomicAdd(o + 0, v.x * norm);
    atomicAdd(o + 1, v.y * norm);
    atomicAdd(o + 2, v.z * norm);
    atomicAdd(o + 3, v.w * norm);
}

// ---------------------------------------------------------------------------
// Fused: hpre = agg + xw*dinv^2 (self loop) + b  (in-place into xw buffer)
//        + BatchNorm column sum / sumsq accumulation (register partials).
// Column-mapped threads, grid-stride over rows.
// ---------------------------------------------------------------------------
__global__ void k_finalize_stats(float* __restrict__ xw, const float* __restrict__ agg,
                                 const float* __restrict__ dinv, const float* __restrict__ b,
                                 float* __restrict__ stats, int nrows) {
    int f = threadIdx.x & (HF - 1);
    float bf = b[f];
    float s = 0.0f, s2 = 0.0f;
    for (int r = blockIdx.x * 2 + (threadIdx.x >> 7); r < nrows; r += gridDim.x * 2) {
        long i = (long)r * HF + f;
        float di = dinv[r];
        float v = agg[i] + xw[i] * di * di + bf;
        xw[i] = v;
        s += v; s2 += v * v;
    }
    atomicAdd(&stats[f], s);
    atomicAdd(&stats[HF + f], s2);
}

__global__ void k_bn_apply(const float* __restrict__ h, const float* __restrict__ stats,
                           const float* __restrict__ g, const float* __restrict__ be,
                           float* __restrict__ xo, int nrows) {
    long i = (long)blockIdx.x * blockDim.x + threadIdx.x;
    if (i >= (long)nrows * HF) return;
    int f = (int)(i & (HF - 1));
    float invn = 1.0f / (float)nrows;
    float mu  = stats[f] * invn;
    float var = stats[HF + f] * invn - mu * mu;
    float y = (h[i] - mu) * rsqrtf(var + EPSBN) * g[f] + be[f];
    xo[i] = y > 0.0f ? y : 0.0f;
}

// ---------------------------------------------------------------------------
// Global mean pool + classifier
// ---------------------------------------------------------------------------
__global__ void k_pool(const float* __restrict__ x, const int* __restrict__ batch,
                       float* __restrict__ pooled, float* __restrict__ cnt, int nrows) {
    int v = __builtin_amdgcn_readfirstlane(blockIdx.x * 8 + (threadIdx.x >> 5));
    if (v >= nrows) return;
    int lane = threadIdx.x & 31;
    int g = batch[v];
    const float4 t = *(const float4*)(x + (long)v * HF + lane * 4);
    float* p = pooled + (long)g * HF + lane * 4;
    atomicAdd(p + 0, t.x); atomicAdd(p + 1, t.y);
    atomicAdd(p + 2, t.z); atomicAdd(p + 3, t.w);
    if (lane == 0) atomicAdd(&cnt[g], 1.0f);
}

__global__ void k_classify(const float* __restrict__ pooled, const float* __restrict__ cnt,
                           const float* __restrict__ Wc, const float* __restrict__ bc,
                           float* __restrict__ out, int ngraphs, int ncls) {
    int t = blockIdx.x * blockDim.x + threadIdx.x;
    if (t >= ngraphs * ncls) return;
    int g = t / ncls, c = t % ncls;
    float inv = 1.0f / fmaxf(cnt[g], 1.0f);
    float acc = bc[c];
    for (int k = 0; k < HF; ++k)
        acc += pooled[(long)g * HF + k] * inv * Wc[k * ncls + c];
    out[t] = acc;
}

// ---------------------------------------------------------------------------
// Host side
// ---------------------------------------------------------------------------
static inline size_t alignup(size_t x) { return (x + 255) & ~(size_t)255; }

extern "C" void kernel_launch(void* const* d_in, const int* in_sizes, int n_in,
                              void* d_out, int out_size, void* d_ws, size_t ws_size,
                              hipStream_t stream) {
    const float* x     = (const float*)d_in[0];
    const int*   eidx  = (const int*)d_in[1];
    const int*   batch = (const int*)d_in[2];
    const float* W[3]  = { (const float*)d_in[3],  (const float*)d_in[7],  (const float*)d_in[11] };
    const float* b[3]  = { (const float*)d_in[4],  (const float*)d_in[8],  (const float*)d_in[12] };
    const float* gm[3] = { (const float*)d_in[5],  (const float*)d_in[9],  (const float*)d_in[13] };
    const float* be[3] = { (const float*)d_in[6],  (const float*)d_in[10], (const float*)d_in[14] };
    const float* Wc = (const float*)d_in[15];
    const float* bc = (const float*)d_in[16];
    float* out = (float*)d_out;

    const int N    = in_sizes[0] / HF;
    const int Npad = (N + 127) & ~127;     // padded rows: unguarded GEMM stores
    const int E    = in_sizes[1] / 2;
    const int C    = in_sizes[16];         // 10
    const int G    = out_size / C;         // 128
    const int* src = eidx;
    const int* dst = eidx + E;

    // workspace carve-out (node buffers padded to Npad rows)
    char* ws = (char*)d_ws;
    size_t off = 0;
    float* xw     = (float*)(ws + off); off = alignup(off + (size_t)Npad * HF * 4);
    float* agg    = (float*)(ws + off); off = alignup(off + (size_t)Npad * HF * 4);
    float* xbuf   = (float*)(ws + off); off = alignup(off + (size_t)Npad * HF * 4);
    float* dinv   = (float*)(ws + off); off = alignup(off + (size_t)N * 4);
    float* stats  = (float*)(ws + off); off = alignup(off + 2 * HF * 4);
    float* pooled = (float*)(ws + off); off = alignup(off + (size_t)G * HF * 4);
    float* cnt    = (float*)(ws + off); off = alignup(off + (size_t)G * 4);
    __bf16* Wswz  = (__bf16*)(ws + off); off = alignup(off + (size_t)3 * HF * HF * 2);
    (void)ws_size; (void)n_in;

    const int T = 256;
    // degrees (once)
    k_deg_init<<<(N + T - 1) / T, T, 0, stream>>>(dinv, N);
    k_deg_edges<<<(E + T - 1) / T, T, 0, stream>>>(dst, dinv, E);
    k_deg_fin<<<(N + T - 1) / T, T, 0, stream>>>(dinv, N);

    // weight swizzle (once per call)
    for (int l = 0; l < 3; ++l)
        k_wswz<<<4, T, 0, stream>>>(W[l], Wswz + (size_t)l * HF * HF);

    const long NH  = (long)N * HF;
    const long NH4 = NH / 4;               // HF divisible by 4
    const int gemmBlocks = Npad / 128;
    const int elemBlocks = (int)((NH + T - 1) / T);
    const int zero4Blocks = (int)((NH4 + T - 1) / T);

    const float* cur = x;
    for (int l = 0; l < 3; ++l) {
        k_gemm<<<gemmBlocks, T, 0, stream>>>(cur, Wswz + (size_t)l * HF * HF, xw, N);
        k_zero4<<<zero4Blocks, T, 0, stream>>>((float4*)agg, NH4);
        k_scatter<<<(E + 7) / 8, T, 0, stream>>>(xw, src, dst, dinv, agg, E);
        k_zero<<<1, T, 0, stream>>>(stats, 2 * HF);
        k_finalize_stats<<<512, T, 0, stream>>>(xw, agg, dinv, b[l], stats, N);
        k_bn_apply<<<elemBlocks, T, 0, stream>>>(xw, stats, gm[l], be[l], xbuf, N);
        cur = xbuf;
    }

    // pooling + classifier
    k_zero4<<<(G * HF / 4 + T - 1) / T, T, 0, stream>>>((float4*)pooled, (long)G * HF / 4);
    k_zero<<<1, T, 0, stream>>>(cnt, G);
    k_pool<<<(N + 7) / 8, T, 0, stream>>>(xbuf, batch, pooled, cnt, N);
    k_classify<<<(G * C + T - 1) / T, T, 0, stream>>>(pooled, cnt, Wc, bc, out, G, C);
}
